// FinalOpOnEdge_69312182223242
// MI455X (gfx1250) — matrix operand order repaired and checked by reading, not compiled, not run on hardware
//
#include <hip/hip_runtime.h>

// ---------------------------------------------------------------------------
// Grouped linear + edge aggregation as one GEMM:
//   out[B,128] = X[B, 4*128] @ W[4*128, 128] + sum_e bs[e]
// bf16x3 split-precision WMMA (V_WMMA_F32_16X16X32_BF16), fp32 accumulate.
// ---------------------------------------------------------------------------

typedef __attribute__((ext_vector_type(16))) __bf16 v16bf;
typedef __attribute__((ext_vector_type(8)))  float  v8f;
typedef __attribute__((ext_vector_type(4)))  float  f32x4;

#define E_ 4
#define B_ 262144
#define C_ 128
#define ROWS_PER_BLOCK 128   // 8 waves x 16 rows
#define THREADS 256

// round-to-nearest-even float -> bf16 bits
static __device__ __forceinline__ unsigned short bf16_rne_bits(float f) {
    unsigned u = __builtin_bit_cast(unsigned, f);
    unsigned r = u + 0x7FFFu + ((u >> 16) & 1u);
    return (unsigned short)(r >> 16);
}
static __device__ __forceinline__ __bf16 bits_to_bf16(unsigned short s) {
    return __builtin_bit_cast(__bf16, s);
}

__global__ __launch_bounds__(THREADS)
void grouped_linear_sum_kernel(const float* __restrict__ xs,   // [E,B,C]
                               const float* __restrict__ Ws,   // [E,C,C]
                               const float* __restrict__ bs,   // [E,C]
                               float* __restrict__ out)        // [B,C]
{
    // W fragments in WMMA-B lane layout, bf16 hi/lo split.
    // Fragment index f = (e*4 + ks)*8 + n  (ks: K-step of 32, n: 16-col tile)
    // Within a fragment: lane*16 + j  (j = 16 bf16 elements of that lane)
    __shared__ __align__(32) unsigned short Whi[65536];   // 128 KB
    __shared__ __align__(32) unsigned short Wlo[65536];   // 128 KB

    const int tid = threadIdx.x;

    // ---- Stage W: coalesced global reads, scatter into fragment layout ----
    // Linear over (e, K, N): consecutive tid -> consecutive N (coalesced).
    #pragma unroll 4
    for (int i = 0; i < (E_ * C_ * C_) / THREADS; ++i) {
        const int g  = i * THREADS + tid;
        const int e  = g >> 14;            // /16384
        const int K  = (g >> 7) & 127;
        const int N  = g & 127;

        const float w = Ws[g];
        const unsigned u = __builtin_bit_cast(unsigned, w);
        const unsigned short hi = (unsigned short)(u >> 16);          // truncation split
        const float hif = __builtin_bit_cast(float, u & 0xFFFF0000u); // exact hi as f32
        const unsigned short lo = bf16_rne_bits(w - hif);             // residual

        // B-matrix 32x16 layout (16-bit): lanes 0-15: N=lane, K=k0+{0..7,16..23};
        //                                 lanes 16-31: N=lane-16, K=k0+{8..15,24..31}
        const int ks   = K >> 5;
        const int kr   = K & 31;
        const int lane = (N & 15) + (((kr >> 3) & 1) ? 16 : 0);
        const int j    = (kr & 7) | ((kr >> 4) << 3);
        const int n    = N >> 4;
        const int lidx = (((e * 4 + ks) * 8 + n) << 9) + (lane << 4) + j;
        Whi[lidx] = hi;
        Wlo[lidx] = lo;
    }
    __syncthreads();

    const int wave = tid >> 5;
    const int lane = tid & 31;
    const int lm   = lane & 15;   // lane % 16
    const int lh   = lane >> 4;   // 0 or 1

    // A-matrix row this lane supplies (A 16-bit layout: lane%16 = M)
    const int row = blockIdx.x * ROWS_PER_BLOCK + wave * 16 + lm;

    v8f acc[8];
    #pragma unroll
    for (int n = 0; n < 8; ++n)
        #pragma unroll
        for (int k = 0; k < 8; ++k) acc[n][k] = 0.0f;

    // ---- Main loop: K = 512 total, 4 edges x 4 K-steps of 32 ----
    for (int e = 0; e < E_; ++e) {
        const float* ap = xs + ((size_t)e * B_ + row) * C_;
        for (int ks = 0; ks < 4; ++ks) {
            const int k0 = ks * 32;
            // A 16-bit layout: lanes 0-15 need K = k0+{0..7, 16..23},
            //                  lanes 16-31 need K = k0+{8..15, 24..31}
            const int off = lh * 8;
            const f32x4 r0 = *(const f32x4*)(ap + k0 + off);
            const f32x4 r1 = *(const f32x4*)(ap + k0 + off + 4);
            const f32x4 r2 = *(const f32x4*)(ap + k0 + 16 + off);
            const f32x4 r3 = *(const f32x4*)(ap + k0 + 16 + off + 4);

            float va[16];
            #pragma unroll
            for (int i = 0; i < 4; ++i) {
                va[i]      = r0[i];
                va[4 + i]  = r1[i];
                va[8 + i]  = r2[i];
                va[12 + i] = r3[i];
            }

            v16bf ahi, alo;
            #pragma unroll
            for (int i = 0; i < 16; ++i) {
                const unsigned u = __builtin_bit_cast(unsigned, va[i]);
                const unsigned short h = (unsigned short)(u >> 16);
                const float hif = __builtin_bit_cast(float, u & 0xFFFF0000u);
                ahi[i] = bits_to_bf16(h);
                alo[i] = bits_to_bf16(bf16_rne_bits(va[i] - hif));
            }

            const int fbase = (((e * 4 + ks) * 8) << 9) + (lane << 4);
            #pragma unroll
            for (int n = 0; n < 8; ++n) {
                const v16bf bhi = *(const v16bf*)(&Whi[fbase + (n << 9)]);
                const v16bf blo = *(const v16bf*)(&Wlo[fbase + (n << 9)]);
                // D = A*B + C ; bf16x3: hi*hi + lo*hi + hi*lo
                acc[n] = __builtin_amdgcn_wmma_f32_16x16x32_bf16(
                            false, ahi, false, bhi, (short)0, acc[n], false, false);
                acc[n] = __builtin_amdgcn_wmma_f32_16x16x32_bf16(
                            false, alo, false, bhi, (short)0, acc[n], false, false);
                acc[n] = __builtin_amdgcn_wmma_f32_16x16x32_bf16(
                            false, ahi, false, blo, (short)0, acc[n], false, false);
            }
        }
    }

    // ---- Epilogue: bias (summed over edges) + store ----
    // C/D layout: VGPR v, lanes 0-15 -> (M=v, N=lane); lanes 16-31 -> (M=v+8, N=lane-16)
    float bsum[8];
    #pragma unroll
    for (int n = 0; n < 8; ++n) {
        const int col = n * 16 + lm;
        float s = 0.0f;
        #pragma unroll
        for (int e = 0; e < E_; ++e) s += bs[e * C_ + col];
        bsum[n] = s;
    }

    const int rbase = blockIdx.x * ROWS_PER_BLOCK + wave * 16 + lh * 8;
    #pragma unroll
    for (int n = 0; n < 8; ++n) {
        const int col = n * 16 + lm;
        #pragma unroll
        for (int v = 0; v < 8; ++v) {
            out[(size_t)(rbase + v) * C_ + col] = acc[n][v] + bsum[n];
        }
    }
}

extern "C" void kernel_launch(void* const* d_in, const int* in_sizes, int n_in,
                              void* d_out, int out_size, void* d_ws, size_t ws_size,
                              hipStream_t stream) {
    const float* xs = (const float*)d_in[0];   // [E,B,C] f32
    const float* Ws = (const float*)d_in[1];   // [E,C,C] f32
    const float* bs = (const float*)d_in[2];   // [E,C]   f32
    float* out = (float*)d_out;                // [B,C]   f32

    dim3 grid(B_ / ROWS_PER_BLOCK);            // 2048 blocks
    dim3 block(THREADS);                       // 8 waves
    hipLaunchKernelGGL(grouped_linear_sum_kernel, grid, block, 0, stream,
                       xs, Ws, bs, out);
}
